// MambaBlock_3435973836964
// MI455X (gfx1250) — compile-verified, matrix operand order checked by reading, not run on hardware
//
#include <hip/hip_runtime.h>

// ---------------- problem constants ----------------
#define B_SZ   4
#define L_SZ   2048
#define DM     1024           // d_model
#define DI     2048           // d_inner
#define DS     64             // d_state
#define DTRANK 64
#define NROW   (B_SZ * L_SZ)  // 8192 rows

typedef __attribute__((ext_vector_type(16))) __bf16 v16bf;
typedef __attribute__((ext_vector_type(8)))  float  v8f;

union FragAB { v16bf v; uint4 q[2]; };

__device__ __forceinline__ unsigned short f2bf(float f) {
  unsigned u = __builtin_bit_cast(unsigned, f);
  u += 0x7fffu + ((u >> 16) & 1u);            // round-to-nearest-even
  return (unsigned short)(u >> 16);
}
__device__ __forceinline__ float sigmoidf_(float v) { return 1.f / (1.f + __expf(-v)); }
__device__ __forceinline__ float softplusf_(float v) {
  return v > 0.f ? v + log1pf(__expf(-v)) : log1pf(__expf(v));
}

// ---- CDNA5 async DMA: global -> LDS, 16B per lane, tracked by ASYNCcnt ----
__device__ __forceinline__ void async_b128(unsigned lds_addr, unsigned long long gaddr) {
  asm volatile("global_load_async_to_lds_b128 %0, %1, off"
               :: "v"(lds_addr), "v"(gaddr) : "memory");
}
__device__ __forceinline__ void wait_async0() {
  asm volatile("s_wait_asynccnt 0x0" ::: "memory");
}
__device__ __forceinline__ unsigned lds_off(const void* p) {  // LDS aperture: low 32 bits
  return (unsigned)(unsigned long long)p;
}

// ---------------- fp32 -> bf16, transposed (weights -> k-major [N][K]) ----
__global__ void k_f32_to_bf16_T(const float* __restrict__ in, unsigned short* __restrict__ out,
                                int K, int N) {   // in: [K][N] row-major
  int i = blockIdx.x * 256 + threadIdx.x;
  if (i < K * N) {
    int k = i / N, n = i % N;
    out[(size_t)n * K + k] = f2bf(in[i]);
  }
}

// ---------------- LayerNorm -> bf16 ----------------
__global__ __launch_bounds__(256) void k_layernorm_bf16(
    const float* __restrict__ x, const float* __restrict__ g,
    const float* __restrict__ be, unsigned short* __restrict__ xn) {
  __shared__ float red[256];
  int row = blockIdx.x, tid = threadIdx.x;
  const float* xr = x + (size_t)row * DM;
  float v[4], s = 0.f, s2 = 0.f;
#pragma unroll
  for (int j = 0; j < 4; ++j) { v[j] = xr[tid + 256 * j]; s += v[j]; s2 += v[j] * v[j]; }
  red[tid] = s; __syncthreads();
  for (int st = 128; st > 0; st >>= 1) { if (tid < st) red[tid] += red[tid + st]; __syncthreads(); }
  float mean = red[0] * (1.f / DM); __syncthreads();
  red[tid] = s2; __syncthreads();
  for (int st = 128; st > 0; st >>= 1) { if (tid < st) red[tid] += red[tid + st]; __syncthreads(); }
  float var = red[0] * (1.f / DM) - mean * mean;
  float rstd = rsqrtf(var + 1e-5f);
#pragma unroll
  for (int j = 0; j < 4; ++j) {
    int c = tid + 256 * j;
    xn[(size_t)row * DM + c] = f2bf((v[j] - mean) * rstd * g[c] + be[c]);
  }
}

// ---------------- bf16 WMMA GEMM: C[MxN] = A[MxK] * Bt[NxK]^T (+epilogue) --
// A row-major [M][K]; Bt k-major [N][K] (pre-transposed weights).
// BM=128; BN in {64,128}. 8 waves: 4 along M (32 rows), 2 along N.
// Double-buffered LDS filled by GLOBAL_LOAD_ASYNC_TO_LDS_B128 (ASYNCcnt).
// MODE 0: plain | MODE 1: softplus(acc + bias[n]) | MODE 2: acc + res[m*N+n]
template <int MODE, int BN>
__global__ __launch_bounds__(256, 1) void k_gemm_bf16(
    const unsigned short* __restrict__ A, const unsigned short* __restrict__ Bt,
    float* __restrict__ C, int M, int N, int K,
    const float* __restrict__ bias, const float* __restrict__ res) {
  constexpr int BM  = 128;
  constexpr int NAV = BM / 64;      // 16B async chunks per thread for A tile
  constexpr int NBV = BN / 64;      // 16B async chunks per thread for B tile
  constexpr int WC  = BN / 2;       // columns per wave
  constexpr int NI  = WC / 16;      // B fragments per wave

  __shared__ __align__(16) unsigned short sA[2][BM * 32];  // [row][k]
  __shared__ __align__(16) unsigned short sB[2][BN * 32];  // [col][k]

  const int tid = threadIdx.x;
  const int lane = tid & 31, wid = tid >> 5;
  const int wm = wid >> 1, wn = wid & 1;
  const int half = lane >> 4, l15 = lane & 15;
  const int row0 = blockIdx.y * BM, col0 = blockIdx.x * BN;

  v8f zero = {};
  v8f acc[2][NI];
#pragma unroll
  for (int i = 0; i < 2; ++i)
#pragma unroll
    for (int j = 0; j < NI; ++j) acc[i][j] = zero;

  // per-thread tile element coordinates (8 bf16 = 16 bytes per chunk)
  // chunk j: e = (tid + j*256)*8 ; r = e>>5 ; kk = e&31
#define ISSUE_TILE(buf, kbase)                                                        \
  {                                                                                   \
    _Pragma("unroll")                                                                 \
    for (int j = 0; j < NAV; ++j) {                                                   \
      int e = (tid + j * 256) * 8, r = e >> 5, kk = e & 31;                           \
      async_b128(lds_off(&sA[buf][r * 32 + kk]),                                      \
                 (unsigned long long)(A + (size_t)(row0 + r) * K + (kbase) + kk));    \
    }                                                                                 \
    _Pragma("unroll")                                                                 \
    for (int j = 0; j < NBV; ++j) {                                                   \
      int e = (tid + j * 256) * 8, r = e >> 5, kk = e & 31;                           \
      async_b128(lds_off(&sB[buf][r * 32 + kk]),                                      \
                 (unsigned long long)(Bt + (size_t)(col0 + r) * K + (kbase) + kk));   \
    }                                                                                 \
  }

  // ---- prologue: DMA tile 0 into buffer 0 ----
  ISSUE_TILE(0, 0)
  wait_async0();
  __syncthreads();

  int cur = 0;
  for (int k0 = 0; k0 < K; k0 += 32) {
    const bool more = (k0 + 32) < K;
    if (more) {
      ISSUE_TILE(cur ^ 1, k0 + 32)                 // DMA next tile while we compute
      if (k0 + 64 < K) {                           // L2 prefetch two tiles ahead
        __builtin_prefetch(A + (size_t)(row0 + (tid >> 1)) * K + k0 + 64, 0, 0);
        __builtin_prefetch(Bt + (size_t)(col0 + (tid & (BN - 1))) * K + k0 + 64, 0, 0);
      }
    }

    // ---- fragments (all ds_load_b128) ----
    FragAB fa[2];
#pragma unroll
    for (int i = 0; i < 2; ++i) {
      const unsigned short* p = &sA[cur][(32 * wm + 16 * i + l15) * 32];
      fa[i].q[0] = *(const uint4*)(p + half * 8);
      fa[i].q[1] = *(const uint4*)(p + 16 + half * 8);
    }
    FragAB fb[NI];
#pragma unroll
    for (int j = 0; j < NI; ++j) {
      const unsigned short* p = &sB[cur][(WC * wn + 16 * j + l15) * 32];
      fb[j].q[0] = *(const uint4*)(p + half * 8);
      fb[j].q[1] = *(const uint4*)(p + 16 + half * 8);
    }
#pragma unroll
    for (int i = 0; i < 2; ++i)
#pragma unroll
      for (int j = 0; j < NI; ++j)
        acc[i][j] = __builtin_amdgcn_wmma_f32_16x16x32_bf16(
            false, fa[i].v, false, fb[j].v, (short)0, acc[i][j], false, false);

    if (more) wait_async0();                       // DMA for next buffer done
    __syncthreads();
    cur ^= 1;
  }
#undef ISSUE_TILE

  // ---- epilogue: D layout VGPR r -> M = r + 8*half, N = lane&15 ----
#pragma unroll
  for (int i = 0; i < 2; ++i)
#pragma unroll
    for (int j = 0; j < NI; ++j)
#pragma unroll
      for (int r = 0; r < 8; ++r) {
        int m_g = row0 + 32 * wm + 16 * i + 8 * half + r;
        int n_g = col0 + WC * wn + 16 * j + l15;
        float v = acc[i][j][r];
        if (MODE == 1) v = softplusf_(v + bias[n_g]);
        size_t o = (size_t)m_g * N + n_g;
        if (MODE == 2) v += res[o];
        C[o] = v;
      }
}

// ---------------- causal depthwise conv (k=4) + SiLU ----------------
__global__ __launch_bounds__(256) void k_conv_silu(
    const float* __restrict__ xz, const float* __restrict__ cw, const float* __restrict__ cb,
    float* __restrict__ xc, unsigned short* __restrict__ xcb) {
  int idx = blockIdx.x * 256 + threadIdx.x;        // over NROW*DI
  int d = idx & (DI - 1);
  int row = idx >> 11;                              // DI = 2^11
  int l = row & (L_SZ - 1);
  const float* w = cw + d * 4;
  float acc = cb[d];
#pragma unroll
  for (int j = 0; j < 4; ++j) {
    int ls = l - 3 + j;
    if (ls >= 0) acc += w[j] * xz[(size_t)(row - 3 + j) * (2 * DI) + d];
  }
  float o = acc * sigmoidf_(acc);
  xc[idx] = o;
  xcb[idx] = f2bf(o);
}

// ---------------- extract dt_rank columns of x_dbl as bf16 ----------------
__global__ void k_extract_dtr(const float* __restrict__ xdbl, unsigned short* __restrict__ dtr) {
  int i = blockIdx.x * 256 + threadIdx.x;          // NROW*64
  int row = i >> 6, c = i & 63;
  dtr[i] = f2bf(xdbl[(size_t)row * 192 + c]);
}

// ---------------- selective scan: 4 threads per (b,d), 16 states each ------
__global__ __launch_bounds__(256) void k_scan(
    const float* __restrict__ dt, const float* __restrict__ xc,
    const float* __restrict__ xdbl, const float* __restrict__ Alog,
    const float* __restrict__ Dskip, const float* __restrict__ xz,
    unsigned short* __restrict__ ybf) {
  __shared__ float Bs[DS], Cs[DS];
  const int tid = threadIdx.x;
  const int gid = blockIdx.x * 256 + tid;
  const int q = gid & 3;                 // state quarter: n in [16q, 16q+16)
  const int pair = gid >> 2;             // (b,d) pair
  const int d = pair & (DI - 1);
  const int b = pair >> 11;
  float a[16], h[16];
#pragma unroll
  for (int j = 0; j < 16; ++j) {
    a[j] = -__expf(Alog[d * DS + 16 * q + j]);
    h[j] = 0.f;
  }
  const float dsk = Dskip[d];
  for (int l = 0; l < L_SZ; ++l) {
    const int row = b * L_SZ + l;
    if (tid < 2 * DS) {
      float v = xdbl[(size_t)row * 192 + 64 + tid];
      if (tid < DS) Bs[tid] = v; else Cs[tid - DS] = v;
    }
    __syncthreads();
    float dtv = dt[(size_t)row * DI + d];
    float xv  = xc[(size_t)row * DI + d];
    float dtx = dtv * xv;
    float y = 0.f;
#pragma unroll
    for (int j = 0; j < 16; ++j) {
      float dA = __expf(dtv * a[j]);
      h[j] = dA * h[j] + dtx * Bs[16 * q + j];
      y += h[j] * Cs[16 * q + j];
    }
    y += __shfl_xor(y, 1);
    y += __shfl_xor(y, 2);
    if (q == 0) {
      float zv = xz[(size_t)row * (2 * DI) + DI + d];
      float yo = (y + xv * dsk) * (zv * sigmoidf_(zv));
      ybf[(size_t)row * DI + d] = f2bf(yo);
    }
    __syncthreads();
  }
}

// ---------------- host orchestration ----------------
extern "C" void kernel_launch(void* const* d_in, const int* in_sizes, int n_in,
                              void* d_out, int out_size, void* d_ws, size_t ws_size,
                              hipStream_t stream) {
  const float* x      = (const float*)d_in[0];
  const float* ln_g   = (const float*)d_in[1];
  const float* ln_b   = (const float*)d_in[2];
  const float* W_in   = (const float*)d_in[3];
  const float* conv_w = (const float*)d_in[4];
  const float* conv_b = (const float*)d_in[5];
  const float* W_x    = (const float*)d_in[6];
  const float* W_dt   = (const float*)d_in[7];
  const float* b_dt   = (const float*)d_in[8];
  const float* A_log  = (const float*)d_in[9];
  const float* Dskip  = (const float*)d_in[10];
  const float* W_out  = (const float*)d_in[11];
  float* out = (float*)d_out;

  char* base = (char*)d_ws;
  size_t off = 0;
  auto alloc = [&](size_t bytes) -> char* {
    char* p = base + off;
    off = (off + bytes + 255) & ~(size_t)255;
    return p;
  };
  unsigned short* xn_b   = (unsigned short*)alloc((size_t)NROW * DM * 2);        // 16 MB
  unsigned short* WinT   = (unsigned short*)alloc((size_t)DM * 2 * DI * 2);      //  8 MB  [4096][1024]
  unsigned short* WxT    = (unsigned short*)alloc((size_t)DI * 192 * 2);         //        [192][2048]
  unsigned short* WdtT   = (unsigned short*)alloc((size_t)DTRANK * DI * 2);      //        [2048][64]
  unsigned short* WoutT  = (unsigned short*)alloc((size_t)DI * DM * 2);          //  4 MB  [1024][2048]
  float*          xz     = (float*)alloc((size_t)NROW * 2 * DI * 4);             // 128 MB
  float*          xc     = (float*)alloc((size_t)NROW * DI * 4);                 // 64 MB
  unsigned short* xc_b   = (unsigned short*)alloc((size_t)NROW * DI * 2);        // 32 MB
  float*          xdbl   = (float*)alloc((size_t)NROW * 192 * 4);                //  6 MB
  unsigned short* dtr_b  = (unsigned short*)alloc((size_t)NROW * DTRANK * 2);
  float*          dt     = (float*)alloc((size_t)NROW * DI * 4);                 // 64 MB
  unsigned short* y_b    = (unsigned short*)alloc((size_t)NROW * DI * 2);        // 32 MB
  (void)ws_size; (void)in_sizes; (void)n_in; (void)out_size;

  // weight conversions (fp32 -> bf16, transposed to k-major [N][K])
  k_f32_to_bf16_T<<<(DM * 2 * DI + 255) / 256, 256, 0, stream>>>(W_in, WinT, DM, 2 * DI);
  k_f32_to_bf16_T<<<(DI * 192 + 255) / 256, 256, 0, stream>>>(W_x, WxT, DI, 192);
  k_f32_to_bf16_T<<<(DTRANK * DI + 255) / 256, 256, 0, stream>>>(W_dt, WdtT, DTRANK, DI);
  k_f32_to_bf16_T<<<(DI * DM + 255) / 256, 256, 0, stream>>>(W_out, WoutT, DI, DM);

  // 1. LayerNorm -> bf16 activations
  k_layernorm_bf16<<<NROW, 256, 0, stream>>>(x, ln_g, ln_b, xn_b);
  // 2. xz = xn @ W_in                       [8192x1024] x [1024x4096]
  k_gemm_bf16<0, 128><<<dim3((2 * DI) / 128, NROW / 128), 256, 0, stream>>>(
      xn_b, WinT, xz, NROW, 2 * DI, DM, nullptr, nullptr);
  // 3. depthwise causal conv + SiLU
  k_conv_silu<<<(NROW * DI) / 256, 256, 0, stream>>>(xz, conv_w, conv_b, xc, xc_b);
  // 4. x_dbl = xc @ W_x                     [8192x2048] x [2048x192]  (BN=64)
  k_gemm_bf16<0, 64><<<dim3(192 / 64, NROW / 128), 256, 0, stream>>>(
      xc_b, WxT, xdbl, NROW, 192, DI, nullptr, nullptr);
  // 5. dt low-rank slice -> bf16
  k_extract_dtr<<<(NROW * DTRANK) / 256, 256, 0, stream>>>(xdbl, dtr_b);
  // 6. dt = softplus(dt_r @ W_dt + b_dt)    [8192x64] x [64x2048]
  k_gemm_bf16<1, 128><<<dim3(DI / 128, NROW / 128), 256, 0, stream>>>(
      dtr_b, WdtT, dt, NROW, DI, DTRANK, b_dt, nullptr);
  // 7. selective scan + Dskip + SiLU(z) gate -> y bf16 (4 threads per (b,d))
  k_scan<<<(B_SZ * DI * 4) / 256, 256, 0, stream>>>(dt, xc, xdbl, A_log, Dskip, xz, y_b);
  // 8. out = y @ W_out + residual           [8192x2048] x [2048x1024]
  k_gemm_bf16<2, 128><<<dim3(DM / 128, NROW / 128), 256, 0, stream>>>(
      y_b, WoutT, out, NROW, DM, DI, nullptr, x);
}
